// HLDGNNetwork_64312840290462
// MI455X (gfx1250) — matrix operand order repaired and checked by reading, not compiled, not run on hardware
//
#include <hip/hip_runtime.h>
#include <hip/hip_bf16.h>

// ---------------- problem constants (match reference) ----------------
constexpr int N    = 50000;
constexpr int E    = 800000;
constexpr int ET   = E + N;      // + self loops = 850000
constexpr int G    = 64;
constexpr int FIN  = 128;
constexpr int H    = 4;
constexpr int C    = 32;
constexpr int HC   = 128;
constexpr int MLPH = 128;
constexpr int ACT  = 10;
constexpr float NEG_SLOPE = 0.2f;

// ---------------- workspace layout (bytes) ----------------
constexpr size_t SZ_WPACK = (size_t)2 * 8 * 4 * 32 * 16 * sizeof(unsigned short); // 64 KB
constexpr size_t SZ_X     = (size_t)N * HC * sizeof(float);                       // 25.6 MB
constexpr size_t SZ_EB    = (size_t)ET * H * sizeof(float);                       // 13.6 MB
constexpr size_t OFF_WPACK = 0;
constexpr size_t OFF_XL    = OFF_WPACK + SZ_WPACK;
constexpr size_t OFF_XR    = OFF_XL + SZ_X;
constexpr size_t OFF_EB    = OFF_XR + SZ_X;
// contiguous zero-init region: nmax | denom | agg | gpool
constexpr size_t OFF_NMAX  = OFF_EB + SZ_EB;
constexpr size_t OFF_DEN   = OFF_NMAX + (size_t)N * H * 4;
constexpr size_t OFF_AGG   = OFF_DEN + (size_t)N * H * 4;
constexpr size_t OFF_GPOOL = OFF_AGG + SZ_X;
constexpr size_t ZERO_DWORDS = (size_t)N * H * 2 + (size_t)N * HC + (size_t)G * HC;

// ---------------- WMMA types ----------------
typedef __attribute__((ext_vector_type(16))) __bf16 v16bf;
typedef __attribute__((ext_vector_type(8)))  float  v8f;

union Frag16 { v16bf v; uint4 q[2]; unsigned short u[16]; };

__device__ __forceinline__ unsigned short f32_to_bf16_rne(float f) {
    unsigned u = __float_as_uint(f);
    unsigned r = (u + 0x7FFFu + ((u >> 16) & 1u)) >> 16;
    return (unsigned short)r;
}
// order-preserving float<->uint map (for atomic max over possibly-negative floats)
__device__ __forceinline__ unsigned encf(float f) {
    unsigned u = __float_as_uint(f);
    return (u & 0x80000000u) ? ~u : (u | 0x80000000u);
}
__device__ __forceinline__ float decf(unsigned k) {
    unsigned u = (k & 0x80000000u) ? (k ^ 0x80000000u) : ~k;
    return __uint_as_float(u);
}

// ---------------- kernel 0: zero scratch region ----------------
__global__ void k_zero(unsigned* p, long n) {
    long t = (long)blockIdx.x * blockDim.x + threadIdx.x;
    if (t < n) p[t] = 0u;
}

// ---------------- kernel 1: repack Wl|Wr into WMMA-B bf16 fragments ----------
// pack index = mat*16384 + ntile*2048 + kk*512 + lane*16 + j
// lane n<16 owns col n, K = kk*32 + 0..15 ; lane n+16 owns col n, K = kk*32+16..31
__global__ void k_pack_w(const float* __restrict__ Wl, const float* __restrict__ Wr,
                         unsigned short* __restrict__ wpack) {
    int t = blockIdx.x * blockDim.x + threadIdx.x;
    if (t >= 2 * 8 * 4 * 32 * 16) return;
    int j     = t & 15;
    int lane  = (t >> 4) & 31;
    int kk    = (t >> 9) & 3;
    int ntile = (t >> 11) & 7;
    int mat   = (t >> 14) & 1;
    int n     = lane & 15;
    int khalf = lane >> 4;
    int k     = kk * 32 + khalf * 16 + j;
    int col   = ntile * 16 + n;
    const float* W = mat ? Wr : Wl;
    wpack[t] = f32_to_bf16_rne(W[k * HC + col]);
}

// ---------------- kernel 2: xl = x@Wl+bl, xr = x@Wr+br via WMMA bf16 --------
// grid = N/16 blocks, block = 256 threads = 8 waves (one column tile each,
// serving BOTH Wl and Wr from one shared A fragment staged in LDS).
//
// LDS staging: 256 threads cooperatively convert the 16x128 f32 x-tile into
// bf16 *already in this kernel's A-fragment order, k = 16*half + j* (4 K-steps x 32 lanes x
// 16 elems = 4 KB). Thread t: row m = t&15, kq = t>>4 covers K = kq*8..kq*8+7,
// which maps to one contiguous 16 B run at (kk = kq>>2, lane = m + 16*((kq>>1)&1),
// joff = 8*(kq&1)) -> single b128 LDS store.
__global__ void __launch_bounds__(256)
k_node_gemm(const float* __restrict__ x,
            const unsigned short* __restrict__ wpack,
            const float* __restrict__ bl, const float* __restrict__ br,
            float* __restrict__ xl, float* __restrict__ xr) {
    __shared__ unsigned short afrag_lds[4 * 32 * 16];   // [kk][lane][j]

    const int m0 = blockIdx.x * 16;

    // ---- cooperative A-tile stage: f32 -> bf16 in fragment layout ----
    {
        const int t  = threadIdx.x;
        const int m  = t & 15;
        const int kq = t >> 4;               // 0..15, covers k = kq*8 .. kq*8+7
        const float* src = x + (size_t)(m0 + m) * FIN + kq * 8;
        float4 f0 = *(const float4*)(src);
        float4 f1 = *(const float4*)(src + 4);
        unsigned short u[8];
        u[0] = f32_to_bf16_rne(f0.x); u[1] = f32_to_bf16_rne(f0.y);
        u[2] = f32_to_bf16_rne(f0.z); u[3] = f32_to_bf16_rne(f0.w);
        u[4] = f32_to_bf16_rne(f1.x); u[5] = f32_to_bf16_rne(f1.y);
        u[6] = f32_to_bf16_rne(f1.z); u[7] = f32_to_bf16_rne(f1.w);
        const int kk   = kq >> 2;
        const int q    = kq & 3;
        const int lane = m + 16 * ((q >> 1) & 1);
        const int joff = 8 * (q & 1);
        *(uint4*)(afrag_lds + kk * 512 + lane * 16 + joff) = *(const uint4*)u;
    }
    __syncthreads();

    const int wave  = threadIdx.x >> 5;      // 0..7 = column tile
    const int lane  = threadIdx.x & 31;
    const int half  = lane >> 4;

    v8f accL = {};
    v8f accR = {};

#pragma unroll
    for (int kk = 0; kk < 4; ++kk) {
        // ---- shared A fragment from LDS (32 B per lane) ----
        Frag16 a;
        const uint4* ap = (const uint4*)(afrag_lds + kk * 512 + lane * 16);
        a.q[0] = ap[0];
        a.q[1] = ap[1];
        // ---- B fragments (pre-packed, 32 contiguous bytes per lane) ----
        Frag16 bL, bR;
        const unsigned short* bpL =
            wpack + (((size_t)(0 * 8 + wave) * 4 + kk) * 32 + lane) * 16;
        const unsigned short* bpR =
            wpack + (((size_t)(1 * 8 + wave) * 4 + kk) * 32 + lane) * 16;
        bL.q[0] = ((const uint4*)bpL)[0];  bL.q[1] = ((const uint4*)bpL)[1];
        bR.q[0] = ((const uint4*)bpR)[0];  bR.q[1] = ((const uint4*)bpR)[1];
        // two independent accumulation chains -> back-to-back XDL issue
        accL = __builtin_amdgcn_wmma_f32_16x16x32_bf16(
            false, a.v, false, bL.v, (short)0, accL, false, false);
        accR = __builtin_amdgcn_wmma_f32_16x16x32_bf16(
            false, a.v, false, bR.v, (short)0, accR, false, false);
    }

    // ---- store per C/D layout: VGPR r -> M = r (+8 for lanes 16-31) ----
    const int col   = wave * 16 + (lane & 15);
    const int rbase = m0 + (half ? 8 : 0);
    const float bvL = bl[col];
    const float bvR = br[col];
#pragma unroll
    for (int r = 0; r < 8; ++r) {
        xl[(size_t)(rbase + r) * HC + col] = accL[r] + bvL;
        xr[(size_t)(rbase + r) * HC + col] = accR[r] + bvR;
    }
}

// ---------------- kernel 3: per-edge GATv2 score + segment max -------------
// thread = (edge, head)
__global__ void k_edge_score(const int* __restrict__ ei,
                             const float* __restrict__ xl, const float* __restrict__ xr,
                             const float* __restrict__ att,
                             float* __restrict__ ebuf, unsigned* __restrict__ nmax) {
    long t = (long)blockIdx.x * blockDim.x + threadIdx.x;
    if (t >= (long)ET * H) return;
    int e = (int)(t >> 2), h = (int)(t & 3);
    int s, d;
    if (e < E) { s = ei[e]; d = ei[E + e]; } else { s = d = e - E; }
    const float4* pl = (const float4*)(xl + (size_t)s * HC + h * C);
    const float4* pr = (const float4*)(xr + (size_t)d * HC + h * C);
    const float4* pa = (const float4*)(att + h * C);
    float acc = 0.f;
#pragma unroll
    for (int j = 0; j < 8; ++j) {
        float4 a4 = pa[j], l4 = pl[j], r4 = pr[j];
        float v;
        v = l4.x + r4.x; v = v > 0.f ? v : NEG_SLOPE * v; acc += a4.x * v;
        v = l4.y + r4.y; v = v > 0.f ? v : NEG_SLOPE * v; acc += a4.y * v;
        v = l4.z + r4.z; v = v > 0.f ? v : NEG_SLOPE * v; acc += a4.z * v;
        v = l4.w + r4.w; v = v > 0.f ? v : NEG_SLOPE * v; acc += a4.w * v;
    }
    ebuf[t] = acc;
    atomicMax(nmax + (size_t)d * H + h, encf(acc));
}

// ---------------- kernel 4: a = exp(e - max), denom += a -------------------
__global__ void k_edge_exp(const int* __restrict__ ei,
                           const unsigned* __restrict__ nmax,
                           float* __restrict__ ebuf, float* __restrict__ denom) {
    long t = (long)blockIdx.x * blockDim.x + threadIdx.x;
    if (t >= (long)ET * H) return;
    int e = (int)(t >> 2), h = (int)(t & 3);
    int d = (e < E) ? ei[E + e] : (e - E);
    float m = decf(nmax[(size_t)d * H + h]);
    float a = __expf(ebuf[t] - m);
    ebuf[t] = a;
    atomicAdd(denom + (size_t)d * H + h, a);
}

// ---------------- kernel 5: agg[dst] += (a/denom) * xl[src] ----------------
__global__ void k_edge_agg(const int* __restrict__ ei,
                           const float* __restrict__ xl,
                           const float* __restrict__ ebuf,
                           const float* __restrict__ denom,
                           float* __restrict__ agg) {
    long t = (long)blockIdx.x * blockDim.x + threadIdx.x;
    if (t >= (long)ET * H) return;
    int e = (int)(t >> 2), h = (int)(t & 3);
    int s, d;
    if (e < E) { s = ei[e]; d = ei[E + e]; } else { s = d = e - E; }
    float alpha = ebuf[t] / denom[(size_t)d * H + h];
    const float* xs = xl + (size_t)s * HC + h * C;
    float* od = agg + (size_t)d * HC + h * C;
#pragma unroll 8
    for (int j = 0; j < C; ++j)
        atomicAdd(od + j, alpha * xs[j]);
}

// ---------------- kernel 6: relu(agg+bias) then per-graph max pool ---------
__global__ void k_pool(const float* __restrict__ agg, const float* __restrict__ cbias,
                       const int* __restrict__ batch, float* __restrict__ gpool) {
    long t = (long)blockIdx.x * blockDim.x + threadIdx.x;
    if (t >= (long)N * HC) return;
    int c = (int)(t & (HC - 1));
    int n = (int)(t >> 7);
    float v = agg[t] + cbias[c];
    v = v > 0.f ? v : 0.f;                          // relu -> non-negative
    int g = batch[n];
    atomicMax((unsigned*)(gpool + (size_t)g * HC + c), __float_as_uint(v));
}

// ---------------- kernel 7: dueling heads on G=64 graphs -------------------
__global__ void __launch_bounds__(128)
k_heads(const float* __restrict__ gpool,
        const float* __restrict__ Wq1, const float* __restrict__ bq1,
        const float* __restrict__ Wq2, const float* __restrict__ bq2,
        const float* __restrict__ Wv1, const float* __restrict__ bv1,
        const float* __restrict__ Wv2, const float* __restrict__ bv2,
        float* __restrict__ out) {
    __shared__ float gl[HC], hq[MLPH], hv[MLPH], qs[ACT + 1];
    const int g = blockIdx.x, j = threadIdx.x;
    gl[j] = gpool[(size_t)g * HC + j];
    __syncthreads();
    float aq = bq1[j], av = bv1[j];
#pragma unroll 8
    for (int k = 0; k < HC; ++k) {
        float gk = gl[k];
        aq += gk * Wq1[k * MLPH + j];
        av += gk * Wv1[k * MLPH + j];
    }
    hq[j] = aq > 0.f ? aq : 0.f;
    hv[j] = av > 0.f ? av : 0.f;
    __syncthreads();
    if (j < ACT) {
        float q = bq2[j];
        for (int k = 0; k < MLPH; ++k) q += hq[k] * Wq2[k * ACT + j];
        qs[j] = q;
    }
    if (j == ACT) {
        float v = bv2[0];
        for (int k = 0; k < MLPH; ++k) v += hv[k] * Wv2[k];
        qs[ACT] = v;
    }
    __syncthreads();
    if (j < ACT) {
        float mean = 0.f;
#pragma unroll
        for (int a = 0; a < ACT; ++a) mean += qs[a];
        mean *= (1.0f / ACT);
        out[(size_t)g * ACT + j] = qs[j] - mean + qs[ACT];
    }
}

// ---------------- host launch ----------------
extern "C" void kernel_launch(void* const* d_in, const int* in_sizes, int n_in,
                              void* d_out, int out_size, void* d_ws, size_t ws_size,
                              hipStream_t stream) {
    const float* x     = (const float*)d_in[0];
    const int*   ei    = (const int*)  d_in[1];
    const int*   batch = (const int*)  d_in[2];
    const float* Wl    = (const float*)d_in[3];
    const float* bl    = (const float*)d_in[4];
    const float* Wr    = (const float*)d_in[5];
    const float* br    = (const float*)d_in[6];
    const float* att   = (const float*)d_in[7];
    const float* cbias = (const float*)d_in[8];
    const float* Wq1   = (const float*)d_in[9];
    const float* bq1   = (const float*)d_in[10];
    const float* Wq2   = (const float*)d_in[11];
    const float* bq2   = (const float*)d_in[12];
    const float* Wv1   = (const float*)d_in[13];
    const float* bv1   = (const float*)d_in[14];
    const float* Wv2   = (const float*)d_in[15];
    const float* bv2   = (const float*)d_in[16];
    float* out = (float*)d_out;

    char* ws = (char*)d_ws;
    unsigned short* wpack = (unsigned short*)(ws + OFF_WPACK);
    float*    xl    = (float*)(ws + OFF_XL);
    float*    xr    = (float*)(ws + OFF_XR);
    float*    ebuf  = (float*)(ws + OFF_EB);
    unsigned* nmax  = (unsigned*)(ws + OFF_NMAX);
    float*    denom = (float*)(ws + OFF_DEN);
    float*    agg   = (float*)(ws + OFF_AGG);
    float*    gpool = (float*)(ws + OFF_GPOOL);

    // 0) zero nmax|denom|agg|gpool (one contiguous region)
    {
        long n = (long)ZERO_DWORDS;
        k_zero<<<(unsigned)((n + 255) / 256), 256, 0, stream>>>((unsigned*)(ws + OFF_NMAX), n);
    }
    // 1) pack weights to bf16 WMMA-B fragments
    k_pack_w<<<(2 * 8 * 4 * 32 * 16) / 256, 256, 0, stream>>>(Wl, Wr, wpack);
    // 2) node transforms via WMMA (LDS-staged shared A fragments)
    k_node_gemm<<<N / 16, 256, 0, stream>>>(x, wpack, bl, br, xl, xr);
    // 3-5) edge phase
    {
        long nt = (long)ET * H;
        unsigned blks = (unsigned)((nt + 255) / 256);
        k_edge_score<<<blks, 256, 0, stream>>>(ei, xl, xr, att, ebuf, nmax);
        k_edge_exp  <<<blks, 256, 0, stream>>>(ei, nmax, ebuf, denom);
        k_edge_agg  <<<blks, 256, 0, stream>>>(ei, xl, ebuf, denom, agg);
    }
    // 6) relu + per-graph max pool
    {
        long nt = (long)N * HC;
        k_pool<<<(unsigned)((nt + 255) / 256), 256, 0, stream>>>(agg, cbias, batch, gpool);
    }
    // 7) dueling heads
    k_heads<<<G, 128, 0, stream>>>(gpool, Wq1, bq1, Wq2, bq2, Wv1, bv1, Wv2, bv2, out);
    (void)in_sizes; (void)n_in; (void)out_size; (void)ws_size;
}
